// ConvIntrinsic_3908420240028
// MI455X (gfx1250) — matrix operand order repaired and checked by reading, not compiled, not run on hardware
//
#include <hip/hip_runtime.h>
#include <hip/hip_bf16.h>

// ConvGeodesic folded into one f16 WMMA GEMM:
//   out[v, o*32+t] = relu( sum_k interp[v,k] * Weff[n,k] + bias[t] )
// with Weff[n=(o,t)][k=(j,f)] = sum_{r,a} coeffs[r,a,j]*weights[t,f,r,(a+2o)%8]
// Shapes: V=100000, R=5, A=8, F=16, T=32, O=4  ->  K=640, N=128.
//
// v3: dual accumulator chains + sched_barrier-enforced A-frag double buffer
//     (forces distinct registers -> partial s_wait_dscnt, LDS latency hidden
//     under WMMA execution). LDS A-tile ping-pong keeps one barrier per tile.

typedef __attribute__((ext_vector_type(16))) _Float16 v16h;
typedef __attribute__((ext_vector_type(8)))  float    v8f;

#define RR 5
#define AA 8
#define FF 16
#define TT 32
#define OO 4
#define KK 640          // R*A*F
#define NN 128          // O*T
#define KSTEPS 20       // 640 / 32
#define LDS_STRIDE 648  // 640 + 8 halfs pad (row stride stays 16B aligned, banks spread)

union F16Frag {
    v16h   v;
    float4 q[2];
};

struct alignas(16) H8 { _Float16 h[8]; };

// ---------------------------------------------------------------------------
// Kernel 1: fold coeffs + rotations into Weff (128 x 640, f16, row-major)
// ---------------------------------------------------------------------------
__global__ void build_weff(const float* __restrict__ weights,  // (T,F,R,A)
                           const float* __restrict__ coeffs,   // (R,A,R*A)
                           _Float16* __restrict__ weff)        // (128,640)
{
    int idx = blockIdx.x * 256 + threadIdx.x;
    if (idx >= NN * KK) return;
    int n = idx / KK;
    int k = idx - n * KK;
    int j = k >> 4;          // 0..39  (= r*8+a template-vertex index)
    int f = k & 15;
    int o = n >> 5;          // orientation
    int t = n & 31;          // template
    int rot = 2 * o;
    float s = 0.f;
#pragma unroll
    for (int r = 0; r < RR; ++r)
#pragma unroll
        for (int a = 0; a < AA; ++a)
            s += coeffs[(r * AA + a) * (RR * AA) + j] *
                 weights[((t * FF + f) * RR + r) * AA + ((a + rot) & 7)];
    weff[n * KK + k] = (_Float16)s;
}

// ---------------------------------------------------------------------------
// Kernel 2: gather + barycentric interp into LDS (f16), then WMMA GEMM.
// 8 waves / block; wave w owns N-tile w and holds its 20 B-fragments (160
// VGPRs) persistently across all vertex tiles.
// ---------------------------------------------------------------------------
__global__ void __launch_bounds__(256)
conv_main(const float* __restrict__ signal,        // (V,16)
          const float* __restrict__ bw,            // (V,40,3)
          const int*   __restrict__ bidx,          // (V,40,3)
          const _Float16* __restrict__ weff,       // (128,640)
          const float* __restrict__ bias,          // (32)
          float* __restrict__ out,                 // (V,128)
          int V, int ntiles)
{
    __shared__ _Float16 Alds[2][16 * LDS_STRIDE];  // 2 x ~20.3 KB ping-pong

    const int tid  = threadIdx.x;
    const int lane = tid & 31;
    const int wave = tid >> 5;       // 0..7 -> N-tile
    const int col  = lane & 15;
    const int h    = lane >> 4;      // lane half

    // ---- persistent B fragments for this wave's N-tile ----
    // B lane layout (16-bit, 32x16): lane holds column n=lane%16,
    // K = {8h..8h+7} and {16+8h..23+8h} -> two b128 loads per K-step.
    const int n = wave * 16 + col;
    const float4* wrow = reinterpret_cast<const float4*>(weff + n * KK);
    float4 Bf[KSTEPS][2];
#pragma unroll
    for (int ks = 0; ks < KSTEPS; ++ks) {
        Bf[ks][0] = wrow[ks * 4 + h];
        Bf[ks][1] = wrow[ks * 4 + 2 + h];
    }
    const float bn = bias[n & 31];

    int p = 0;                       // LDS ping-pong selector
    for (int tile = blockIdx.x; tile < ntiles; tile += gridDim.x) {
        const int vbase = tile * 16;

        // ---- cooperative gather + barycentric interp -> LDS f16 ----
        for (int it = tid; it < 16 * 40; it += 256) {
            const int vv = it / 40;
            const int jj = it - vv * 40;
            int v = vbase + vv;
            if (v >= V) v = V - 1;                       // clamp (tail safety)
            const int base = (v * 40 + jj) * 3;
            const int i0 = bidx[base + 0], i1 = bidx[base + 1], i2 = bidx[base + 2];
            const float w0 = bw[base + 0], w1 = bw[base + 1], w2 = bw[base + 2];
            const float4* s0 = reinterpret_cast<const float4*>(signal + (size_t)i0 * FF);
            const float4* s1 = reinterpret_cast<const float4*>(signal + (size_t)i1 * FF);
            const float4* s2 = reinterpret_cast<const float4*>(signal + (size_t)i2 * FF);
            _Float16* dst = &Alds[p][vv * LDS_STRIDE + jj * FF];
#pragma unroll
            for (int q = 0; q < 2; ++q) {                // two 8-half stores
                H8 pack;
#pragma unroll
                for (int pp = 0; pp < 2; ++pp) {
                    float4 a = s0[q * 2 + pp], b = s1[q * 2 + pp], c = s2[q * 2 + pp];
                    pack.h[pp * 4 + 0] = (_Float16)(w0 * a.x + w1 * b.x + w2 * c.x);
                    pack.h[pp * 4 + 1] = (_Float16)(w0 * a.y + w1 * b.y + w2 * c.y);
                    pack.h[pp * 4 + 2] = (_Float16)(w0 * a.z + w1 * b.z + w2 * c.z);
                    pack.h[pp * 4 + 3] = (_Float16)(w0 * a.w + w1 * b.w + w2 * c.w);
                }
                *reinterpret_cast<H8*>(dst + q * 8) = pack;
            }
        }
        __syncthreads();   // single barrier per tile (ping-pong makes the
                           // write-after-read race impossible: buffer p is
                           // only rewritten two tiles later, after the next
                           // barrier which follows every wave's reads)

        // ---- WMMA: 20 x v_wmma_f32_16x16x32_f16 in two independent
        //      accumulator chains, A-frags double-buffered in pairs.
        //      sched_barrier(0) pins next-pair loads BEFORE current-pair
        //      WMMAs so the buffers' live ranges overlap (no coalescing)
        //      and the waitcnt pass can emit partial s_wait_dscnt. ----
        // A lane layout (16-bit, 16x32): row m = lane%16,
        // K = {8h..8h+7} and {16+8h..23+8h} -> two b128 LDS loads per step.
        v8f acc0 = {}, acc1 = {};
        const float4* arow =
            reinterpret_cast<const float4*>(&Alds[p][col * LDS_STRIDE]);
        F16Frag af[2][2];                                // [buffer][chain]
        af[0][0].q[0] = arow[0 + h];                     // ks = 0
        af[0][0].q[1] = arow[2 + h];
        af[0][1].q[0] = arow[4 + h];                     // ks = 1
        af[0][1].q[1] = arow[6 + h];
#pragma unroll
        for (int kp = 0; kp < KSTEPS; kp += 2) {
            const int cb = (kp >> 1) & 1;                // current buffer
            const int nb = cb ^ 1;                       // next buffer
            if (kp + 2 < KSTEPS) {                       // prefetch next pair
                af[nb][0].q[0] = arow[(kp + 2) * 4 + h];
                af[nb][0].q[1] = arow[(kp + 2) * 4 + 2 + h];
                af[nb][1].q[0] = arow[(kp + 3) * 4 + h];
                af[nb][1].q[1] = arow[(kp + 3) * 4 + 2 + h];
            }
            __builtin_amdgcn_sched_barrier(0);           // loads stay above
            F16Frag b0, b1;
            b0.q[0] = Bf[kp][0];
            b0.q[1] = Bf[kp][1];
            b1.q[0] = Bf[kp + 1][0];
            b1.q[1] = Bf[kp + 1][1];
            acc0 = __builtin_amdgcn_wmma_f32_16x16x32_f16(
                false, af[cb][0].v, false, b0.v, (short)0, acc0, false, false);
            acc1 = __builtin_amdgcn_wmma_f32_16x16x32_f16(
                false, af[cb][1].v, false, b1.v, (short)0, acc1, false, false);
        }
        acc0 = acc0 + acc1;                              // combine chains

        // ---- bias + relu + store ----
        // C/D layout: VGPR d holds row m = d + 8h, column n = lane%16.
#pragma unroll
        for (int d = 0; d < 8; ++d) {
            const int v = vbase + d + 8 * h;
            if (v < V) {
                float val = acc0[d] + bn;
                out[(size_t)v * NN + n] = val > 0.f ? val : 0.f;
            }
        }
        p ^= 1;
    }
}

// ---------------------------------------------------------------------------
// Launch. Input dict order: signal, bary_weights, weights, bias, coeffs,
// bary_indices. Output: (V, 4, 32) f32. Workspace: Weff f16 (160 KB).
// ---------------------------------------------------------------------------
extern "C" void kernel_launch(void* const* d_in, const int* in_sizes, int n_in,
                              void* d_out, int out_size, void* d_ws, size_t ws_size,
                              hipStream_t stream) {
    const float* signal = (const float*)d_in[0];
    const float* bw     = (const float*)d_in[1];
    const float* wts    = (const float*)d_in[2];
    const float* bias   = (const float*)d_in[3];
    const float* coeffs = (const float*)d_in[4];
    const int*   bidx   = (const int*)d_in[5];
    float* out          = (float*)d_out;
    _Float16* weff      = (_Float16*)d_ws;

    const int V = in_sizes[0] / FF;          // 100000
    const int ntiles = (V + 15) / 16;        // 6250

    build_weff<<<(NN * KK + 255) / 256, 256, 0, stream>>>(wts, coeffs, weff);

    int grid = ntiles < 1024 ? ntiles : 1024;
    conv_main<<<grid, 256, 0, stream>>>(signal, bw, bidx, weff, bias, out,
                                        V, ntiles);
}
